// BaseDecoderNetwork_89172110999879
// MI455X (gfx1250) — matrix operand order repaired and checked by reading, not compiled, not run on hardware
//
#include <hip/hip_runtime.h>

#define DEV __device__ __forceinline__

constexpr int B = 512, S = 256, H = 256, V = 32;
constexpr int SMEM_FLOATS = 53760; // dynamic LDS for decoder kernel

typedef __bf16 bf16_t;
typedef __attribute__((ext_vector_type(16))) __bf16 v16bf;
typedef __attribute__((ext_vector_type(8)))  float  v8f;

// ---------- small helpers ----------
DEV bf16_t f2bf(float f) { return (bf16_t)f; }              // native cvt (RNE)
DEV unsigned short f2bf_u(float f) {
  bf16_t b = (bf16_t)f;
  return __builtin_bit_cast(unsigned short, b);
}
DEV float bflo(unsigned u) { return __uint_as_float(u << 16); }
DEV float bfhi(unsigned u) { return __uint_as_float(u & 0xffff0000u); }
DEV float sigm(float x) { return 1.0f / (1.0f + __expf(-x)); }

DEV float wred_max(float v) {
#pragma unroll
  for (int o = 16; o; o >>= 1) v = fmaxf(v, __shfl_xor(v, o, 32));
  return v;
}
DEV float wred_sum(float v) {
#pragma unroll
  for (int o = 16; o; o >>= 1) v += __shfl_xor(v, o, 32);
  return v;
}

// ---------- WMMA fragment packing ----------
// A fragment 16x32 bf16 from fp32 LDS tile (row stride ld floats).
// CDNA5 16-bit A layout: lane m=lane&15; lanes0-15 hold K 0..7,16..23;
// lanes16-31 hold K 8..15,24..31.
DEV v16bf packA_lds(const float* src, int ld, int lane) {
  int m = lane & 15;
  int kb = (lane >> 4) * 8;
  const float* p = src + m * ld + kb;
  float4 f0 = *(const float4*)(p);
  float4 f1 = *(const float4*)(p + 4);
  float4 f2 = *(const float4*)(p + 16);
  float4 f3 = *(const float4*)(p + 20);
  v16bf a;
  a[0] = f2bf(f0.x); a[1] = f2bf(f0.y); a[2] = f2bf(f0.z); a[3] = f2bf(f0.w);
  a[4] = f2bf(f1.x); a[5] = f2bf(f1.y); a[6] = f2bf(f1.z); a[7] = f2bf(f1.w);
  a[8] = f2bf(f2.x); a[9] = f2bf(f2.y); a[10] = f2bf(f2.z); a[11] = f2bf(f2.w);
  a[12] = f2bf(f3.x); a[13] = f2bf(f3.y); a[14] = f2bf(f3.z); a[15] = f2bf(f3.w);
  return a;
}

// B fragment from pre-packed bf16 weights: fragment (kt,nt) stored as
// 32 lanes x 16 bf16 contiguous (32B/lane, fully coalesced uint4 pair).
DEV v16bf loadB_pk(const unsigned short* Wpk, int ntiles, int kt, int nt, int lane) {
  const unsigned short* p = Wpk + (((size_t)(kt * ntiles + nt) * 32 + lane) << 4);
  union { v16bf v; uint4 u[2]; } cv;
  cv.u[0] = *(const uint4*)p;
  cv.u[1] = *(const uint4*)(p + 8);
  return cv.v;
}

// out[16][N] (LDS, stride N) = A1(16xK1)@W1 [+ A2(16xK2)@W2] [+ bias1+bias2]
// ACT: 0=none, 1=tanh. Each participating wave owns exactly NTW N-tiles
// (nt = wave + 8*i), kt-outer loop so one A pack feeds NTW WMMAs.
template <int ACT, int NTW>
DEV void gemm16(const float* A1, int ld1, int K1, const unsigned short* W1,
                const float* A2, int ld2, int K2, const unsigned short* W2,
                const float* bias1, const float* bias2,
                float* out, int N, int wave, int lane) {
  int ntiles = N >> 4;
  if (wave >= ntiles) return;
  v8f zero = {0.f, 0.f, 0.f, 0.f, 0.f, 0.f, 0.f, 0.f};
  v8f acc[NTW];
#pragma unroll
  for (int i = 0; i < NTW; ++i) acc[i] = zero;
  int kt1 = K1 >> 5;
  for (int kt = 0; kt < kt1; ++kt) {
    v16bf a = packA_lds(A1 + kt * 32, ld1, lane);
#pragma unroll
    for (int i = 0; i < NTW; ++i) {
      v16bf b = loadB_pk(W1, ntiles, kt, wave + 8 * i, lane);
      acc[i] = __builtin_amdgcn_wmma_f32_16x16x32_bf16(false, a, false, b,
                                                       (short)0, acc[i], false, false);
    }
  }
  if (A2) {
    int kt2 = K2 >> 5;
    for (int kt = 0; kt < kt2; ++kt) {
      v16bf a = packA_lds(A2 + kt * 32, ld2, lane);
#pragma unroll
      for (int i = 0; i < NTW; ++i) {
        v16bf b = loadB_pk(W2, ntiles, kt, wave + 8 * i, lane);
        acc[i] = __builtin_amdgcn_wmma_f32_16x16x32_bf16(false, a, false, b,
                                                         (short)0, acc[i], false, false);
      }
    }
  }
  int mb = (lane >> 4) * 8;
#pragma unroll
  for (int i = 0; i < NTW; ++i) {
    int n = (wave + 8 * i) * 16 + (lane & 15);
    float bias = 0.f;
    if (bias1) bias += bias1[n];
    if (bias2) bias += bias2[n];
#pragma unroll
    for (int r = 0; r < 8; ++r) {
      float v = acc[i][r] + bias;
      if (ACT == 1) v = tanhf(v);
      out[(size_t)(mb + r) * N + n] = v;
    }
  }
}

// ---------- per-row Luong attention (bf16 GEMV + wave32 softmax) ----------
// q: LDS fp32 [16][256]; dst[r*dstStride + h] <- context.  Wave w owns rows 2w,2w+1.
DEV void attention(const float* q, float* dst, int dstStride,
                   const unsigned short* encP_blk, const unsigned short* enc_blk,
                   float* aw, int wave, int lane) {
#pragma unroll 1
  for (int rr = 0; rr < 2; ++rr) {
    int r = wave * 2 + rr;
    const float* qr = q + r * 256;
    const unsigned short* eP = encP_blk + (size_t)r * S * H;
    float sc[8];
#pragma unroll 1
    for (int j = 0; j < 8; ++j) {
      int s = lane + 32 * j;
      const unsigned short* row = eP + (size_t)s * H;
      if (j < 7) __builtin_prefetch(row + 32 * H, 0, 0);  // next s-row for this lane
      float acc = 0.f;
#pragma unroll 4
      for (int h = 0; h < H; h += 8) {
        uint4 u = *(const uint4*)(row + h);
        float4 q0 = *(const float4*)(qr + h);
        float4 q1 = *(const float4*)(qr + h + 4);
        acc += bflo(u.x) * q0.x + bfhi(u.x) * q0.y
             + bflo(u.y) * q0.z + bfhi(u.y) * q0.w
             + bflo(u.z) * q1.x + bfhi(u.z) * q1.y
             + bflo(u.w) * q1.z + bfhi(u.w) * q1.w;
      }
      sc[j] = acc;
    }
    float m = sc[0];
#pragma unroll
    for (int j = 1; j < 8; ++j) m = fmaxf(m, sc[j]);
    m = wred_max(m);
    float ssum = 0.f;
#pragma unroll
    for (int j = 0; j < 8; ++j) { sc[j] = __expf(sc[j] - m); ssum += sc[j]; }
    ssum = wred_sum(ssum);
    float inv = 1.f / ssum;
#pragma unroll
    for (int j = 0; j < 8; ++j) aw[r * 256 + lane + 32 * j] = sc[j] * inv;

    // weighted sum: lane owns 8 consecutive h
    const unsigned short* eE = enc_blk + (size_t)r * S * H;
    int hb = lane * 8;
    float a0 = 0, a1 = 0, a2 = 0, a3 = 0, a4 = 0, a5 = 0, a6 = 0, a7 = 0;
#pragma unroll 1
    for (int sb = 0; sb < S; sb += 8) {
      __builtin_prefetch(eE + (size_t)(sb + 8) * H + hb, 0, 0);
#pragma unroll
      for (int s2 = 0; s2 < 8; ++s2) {
        int s = sb + s2;
        float w = aw[r * 256 + s];
        uint4 u = *(const uint4*)(eE + (size_t)s * H + hb);
        a0 += w * bflo(u.x); a1 += w * bfhi(u.x);
        a2 += w * bflo(u.y); a3 += w * bfhi(u.y);
        a4 += w * bflo(u.z); a5 += w * bfhi(u.z);
        a6 += w * bflo(u.w); a7 += w * bfhi(u.w);
      }
    }
    float* d = dst + (size_t)r * dstStride + hb;
    d[0] = a0; d[1] = a1; d[2] = a2; d[3] = a3;
    d[4] = a4; d[5] = a5; d[6] = a6; d[7] = a7;
  }
}

// ---------- preprocessing kernels ----------
__global__ void k_bf16(const float* __restrict__ src, unsigned short* __restrict__ dst, size_t n) {
  for (size_t i = (size_t)blockIdx.x * blockDim.x + threadIdx.x; i < n;
       i += (size_t)gridDim.x * blockDim.x)
    dst[i] = f2bf_u(src[i]);
}

__global__ void k_act(const int* __restrict__ a, float* __restrict__ out, int n) {
  for (int i = blockIdx.x * blockDim.x + threadIdx.x; i < n; i += gridDim.x * blockDim.x)
    out[i] = (float)a[i];
}

// Pack W[G][K] fp32 -> bf16 B-fragments (transposed, WMMA lane order).
__global__ void k_pack_w(const float* __restrict__ src, unsigned short* __restrict__ dst,
                         int G, int K) {
  int ntiles = G >> 4;
  size_t total = (size_t)G * K;
  for (size_t i = (size_t)blockIdx.x * blockDim.x + threadIdx.x; i < total;
       i += (size_t)gridDim.x * blockDim.x) {
    size_t f = i >> 9;           // fragment = 512 elems
    int rem = (int)(i & 511);
    int lane = rem >> 4, e = rem & 15;
    int nt = (int)(f % ntiles), kt = (int)(f / ntiles);
    int n = nt * 16 + (lane & 15);
    int k = kt * 32 + ((lane >> 4) << 3) + (e & 7) + ((e >> 3) << 4);
    dst[i] = f2bf_u(src[(size_t)n * K + k]);
  }
}

// enc_proj[b,s,g] = sum_h enc[b,s,h] * W_attn[g,h]   (M=B*S GEMM via WMMA)
__global__ void __launch_bounds__(256) k_encproj(const float* __restrict__ enc,
                                                 const unsigned short* __restrict__ WpkA,
                                                 unsigned short* __restrict__ encPB) {
  __shared__ float As[16 * 256];
  int tid = threadIdx.x, lane = tid & 31, wave = tid >> 5;
  size_t m0 = (size_t)blockIdx.x * 16;
  for (int i = tid; i < 4096; i += 256) As[i] = enc[m0 * 256 + i];
  __syncthreads();
  v8f zero = {0.f, 0.f, 0.f, 0.f, 0.f, 0.f, 0.f, 0.f};
  v8f acc[2] = {zero, zero};
  for (int kt = 0; kt < 8; ++kt) {
    v16bf a = packA_lds(As + kt * 32, 256, lane);
#pragma unroll
    for (int i = 0; i < 2; ++i) {
      v16bf b = loadB_pk(WpkA, 16, kt, wave + 8 * i, lane);
      acc[i] = __builtin_amdgcn_wmma_f32_16x16x32_bf16(false, a, false, b,
                                                       (short)0, acc[i], false, false);
    }
  }
  int mb = (lane >> 4) * 8;
#pragma unroll
  for (int i = 0; i < 2; ++i) {
    int n = (wave + 8 * i) * 16 + (lane & 15);
#pragma unroll
    for (int r = 0; r < 8; ++r)
      encPB[(m0 + mb + r) * 256 + n] = f2bf_u(acc[i][r]);
  }
}

// ---------- persistent decoder: 1 block = 16 batch rows, all 256 steps ----------
__global__ void __launch_bounds__(256, 1) k_decoder(
    const unsigned short* __restrict__ encB, const unsigned short* __restrict__ encPB,
    const unsigned short* __restrict__ Wih0, const unsigned short* __restrict__ Whh0,
    const unsigned short* __restrict__ Wih1, const unsigned short* __restrict__ Whh1,
    const unsigned short* __restrict__ Wcat, const unsigned short* __restrict__ Wout,
    const float* __restrict__ bih0, const float* __restrict__ bhh0,
    const float* __restrict__ bih1, const float* __restrict__ bhh1,
    const float* __restrict__ Wcrit, const float* __restrict__ bcrit,
    const float* __restrict__ h0_in, const float* __restrict__ c0_in,
    const float* __restrict__ embedding, const int* __restrict__ actions,
    float* __restrict__ pi_out, float* __restrict__ val_out) {
  extern __shared__ float sm[];
  float* ctx = sm;             // [16][256]
  float* h0s = sm + 4096;
  float* c0s = sm + 8192;
  float* h1s = sm + 12288;
  float* c1s = sm + 16384;
  float* xb  = sm + 20480;     // [16][512]  emb || attn
  float* gts = sm + 28672;     // [16][1024]
  float* aw  = sm + 45056;     // [16][256]
  float* o2  = sm + 49152;     // [16][256]
  float* lg  = sm + 53248;     // [16][32]

  int tid = threadIdx.x, lane = tid & 31, wave = tid >> 5;
  int b0 = blockIdx.x * 16;
  const unsigned short* encP_blk = encPB + (size_t)b0 * S * H;
  const unsigned short* enc_blk  = encB  + (size_t)b0 * S * H;

  for (int i = tid; i < 4096; i += 256) {
    int r = i >> 8, h = i & 255;
    ctx[i] = 0.f;
    h0s[i] = h0_in[(size_t)(b0 + r) * H + h];
    c0s[i] = c0_in[(size_t)(b0 + r) * H + h];
    h1s[i] = h0_in[(size_t)B * H + (size_t)(b0 + r) * H + h];
    c1s[i] = c0_in[(size_t)B * H + (size_t)(b0 + r) * H + h];
  }

  for (int t = 0; t < S; ++t) {
    __syncthreads();
    // embedding gather into xb[:, :256]
    for (int i = tid; i < 4096; i += 256) {
      int r = i >> 8, h = i & 255;
      int a = actions[(size_t)(b0 + r) * S + t];
      xb[r * 512 + h] = embedding[(size_t)a * H + h];
    }
    // attn(ctx) -> xb[:, 256:512]
    attention(ctx, xb + 256, 512, encP_blk, enc_blk, aw, wave, lane);
    __syncthreads();
    // layer-0 gates: x@W_ih0^T + h0@W_hh0^T + b
    gemm16<0, 8>(xb, 512, 512, Wih0, h0s, 256, 256, Whh0, bih0, bhh0, gts, 1024, wave, lane);
    __syncthreads();
    for (int i = tid; i < 4096; i += 256) {
      int r = i >> 8, j = i & 255;
      const float* g = gts + r * 1024;
      float c = sigm(g[256 + j]) * c0s[i] + sigm(g[j]) * tanhf(g[512 + j]);
      c0s[i] = c;
      h0s[i] = sigm(g[768 + j]) * tanhf(c);
    }
    __syncthreads();
    // layer-1 gates
    gemm16<0, 8>(h0s, 256, 256, Wih1, h1s, 256, 256, Whh1, bih1, bhh1, gts, 1024, wave, lane);
    __syncthreads();
    for (int i = tid; i < 4096; i += 256) {
      int r = i >> 8, j = i & 255;
      const float* g = gts + r * 1024;
      float c = sigm(g[256 + j]) * c1s[i] + sigm(g[j]) * tanhf(g[512 + j]);
      c1s[i] = c;
      h1s[i] = sigm(g[768 + j]) * tanhf(c);
    }
    __syncthreads();
    // attn(out) -> ctx
    attention(h1s, ctx, 256, encP_blk, enc_blk, aw, wave, lane);
    __syncthreads();
    // cat = [h1 || ctx]
    for (int i = tid; i < 4096; i += 256) {
      int r = i >> 8, h = i & 255;
      xb[r * 512 + h] = h1s[i];
      xb[r * 512 + 256 + h] = ctx[i];
    }
    __syncthreads();
    // out2 = tanh(cat @ W_concat^T)
    gemm16<1, 2>(xb, 512, 512, Wcat, nullptr, 0, 0, nullptr, nullptr, nullptr,
                 o2, 256, wave, lane);
    __syncthreads();
    // logits = out2 @ W_out^T   (only waves 0,1 participate)
    gemm16<0, 1>(o2, 256, 256, Wout, nullptr, 0, 0, nullptr, nullptr, nullptr,
                 lg, 32, wave, lane);
    __syncthreads();
    // softmax + critic + outputs (lane = vocab id)
#pragma unroll 1
    for (int rr = 0; rr < 2; ++rr) {
      int r = wave * 2 + rr;
      float l = lg[r * 32 + lane];
      float m = wred_max(l);
      float e = __expf(l - m);
      float ssum = wred_sum(e);
      float p = e / ssum;
      pi_out[((size_t)(b0 + r) * S + t) * V + lane] = p;
      float q = bcrit[lane];
#pragma unroll 8
      for (int k2 = 0; k2 < V; ++k2) q += lg[r * 32 + k2] * Wcrit[lane * V + k2];
      float val = wred_sum(q * p);
      if (lane == 0) val_out[(size_t)(b0 + r) * S + t] = val;
    }
  }
}

// ---------- host ----------
extern "C" void kernel_launch(void* const* d_in, const int* in_sizes, int n_in,
                              void* d_out, int out_size, void* d_ws, size_t ws_size,
                              hipStream_t stream) {
  (void)in_sizes; (void)n_in; (void)out_size; (void)ws_size;
  const float* enc   = (const float*)d_in[0];
  const float* h0    = (const float*)d_in[1];
  const float* c0    = (const float*)d_in[2];
  const float* emb   = (const float*)d_in[3];
  const float* Wattn = (const float*)d_in[4];
  const float* Wih0  = (const float*)d_in[5];
  const float* Whh0  = (const float*)d_in[6];
  const float* bih0  = (const float*)d_in[7];
  const float* bhh0  = (const float*)d_in[8];
  const float* Wih1  = (const float*)d_in[9];
  const float* Whh1  = (const float*)d_in[10];
  const float* bih1  = (const float*)d_in[11];
  const float* bhh1  = (const float*)d_in[12];
  const float* Wcat  = (const float*)d_in[13];
  const float* Wout  = (const float*)d_in[14];
  const float* Wcrit = (const float*)d_in[15];
  const float* bcrit = (const float*)d_in[16];
  const int*   acts  = (const int*)d_in[17];

  char* ws = (char*)d_ws;
  size_t off = 0;
  auto take = [&](size_t bytes) {
    char* p = ws + off;
    off += (bytes + 255) & ~(size_t)255;
    return p;
  };
  unsigned short* encB   = (unsigned short*)take((size_t)B * S * H * 2);
  unsigned short* encPB  = (unsigned short*)take((size_t)B * S * H * 2);
  unsigned short* pA     = (unsigned short*)take((size_t)H * H * 2);          // W_attn
  unsigned short* pIh0   = (unsigned short*)take((size_t)4 * H * 2 * H * 2);  // 1024x512
  unsigned short* pHh0   = (unsigned short*)take((size_t)4 * H * H * 2);
  unsigned short* pIh1   = (unsigned short*)take((size_t)4 * H * H * 2);
  unsigned short* pHh1   = (unsigned short*)take((size_t)4 * H * H * 2);
  unsigned short* pCat   = (unsigned short*)take((size_t)H * 2 * H * 2);      // 256x512
  unsigned short* pOut   = (unsigned short*)take((size_t)V * H * 2);          // 32x256

  float* out      = (float*)d_out;
  float* act_out  = out;                       // [B,S]
  float* pi_out   = out + (size_t)B * S;       // [B,S,V]
  float* val_out  = pi_out + (size_t)B * S * V;

  // preprocessing
  k_bf16<<<4096, 256, 0, stream>>>(enc, encB, (size_t)B * S * H);
  k_pack_w<<<64, 256, 0, stream>>>(Wattn, pA, H, H);
  k_pack_w<<<512, 256, 0, stream>>>(Wih0, pIh0, 4 * H, 2 * H);
  k_pack_w<<<256, 256, 0, stream>>>(Whh0, pHh0, 4 * H, H);
  k_pack_w<<<256, 256, 0, stream>>>(Wih1, pIh1, 4 * H, H);
  k_pack_w<<<256, 256, 0, stream>>>(Whh1, pHh1, 4 * H, H);
  k_pack_w<<<128, 256, 0, stream>>>(Wcat, pCat, H, 2 * H);
  k_pack_w<<<8, 256, 0, stream>>>(Wout, pOut, V, H);
  k_act<<<512, 256, 0, stream>>>(acts, act_out, B * S);
  // enc_proj GEMM (B*S x H x H) in bf16 WMMA
  k_encproj<<<(B * S) / 16, 256, 0, stream>>>(enc, pA, encPB);
  // persistent decoder: 32 blocks x 8 waves, all 256 steps internal
  k_decoder<<<B / 16, 256, SMEM_FLOATS * sizeof(float), stream>>>(
      encB, encPB, pIh0, pHh0, pIh1, pHh1, pCat, pOut,
      bih0, bhh0, bih1, bhh1, Wcrit, bcrit, h0, c0, emb, acts,
      pi_out, val_out);
}
